// UniversalDirectionalGSA_36344013258853
// MI455X (gfx1250) — compile-verified, matrix operand order checked by reading, not run on hardware
//
#include <hip/hip_runtime.h>
#include <hip/hip_bf16.h>

// Problem constants (reference: B,S,HID,NH,NS = 2,1024,2048,16,8; D=128)
#define B_   2
#define S_   1024
#define HID_ 2048
#define NH_  16
#define NS_  8
#define D_   128

typedef __attribute__((ext_vector_type(16))) __bf16 bf16x16;
typedef __attribute__((ext_vector_type(8)))  float  f32x8;

struct Frag { union { bf16x16 v; unsigned int u[8]; }; };

__device__ inline unsigned short f2bf(float f) {
  union { float f; unsigned int u; } v; v.f = f;
  unsigned int u = v.u + (0x7FFFu + ((v.u >> 16) & 1u));  // RNE truncate
  return (unsigned short)(u >> 16);
}
__device__ inline float bf2f(unsigned short h) {
  union { unsigned int u; float f; } v; v.u = ((unsigned int)h) << 16; return v.f;
}
__device__ inline float sigmoidf_(float x) { return 1.0f / (1.0f + __expf(-x)); }

__device__ inline f32x8 wmma_bf16(bf16x16 a, bf16x16 b, f32x8 c) {
  // (neg_a, A, neg_b, B, c_mod, C, reuse_a, reuse_b)
  return __builtin_amdgcn_wmma_f32_16x16x32_bf16(false, a, false, b, (short)0, c,
                                                 false, false);
}

// Async copy of 32 contiguous bytes global -> LDS on the ASYNCcnt path.
// Per the ISA, INST_OFFSET is added to both the LDS and the global address,
// i.e. the immediate offset advances both sides, so one (lds, global) pair
// serves both b128 transfers. Low 32 bits of a generic LDS pointer are the
// LDS byte offset (aperture upper bits are discarded by hardware).
__device__ inline void async_copy32(const unsigned short* gsrc, unsigned short* lds_dst) {
  unsigned lds_off = (unsigned)(size_t)lds_dst;
  asm volatile(
      "global_load_async_to_lds_b128 %0, %1, off\n\t"
      "global_load_async_to_lds_b128 %0, %1, off offset:16"
      :
      : "v"(lds_off), "v"(gsrc)
      : "memory");
}
__device__ inline void wait_async0() {
  asm volatile("s_wait_asynccnt 0x0" ::: "memory");
}

// Load a 16x32 bf16 A/B fragment. Memory layout: the fragment's 16 "rows"
// (M for A, N for B) are contiguous runs of 32 bf16 K-values starting at
// elemBase. ISA pair index for VGPR j: pair = j + ((j>>2)<<2) + half*4.
// Each pair is a 4-byte aligned dword read.
__device__ inline Frag frag_from(const unsigned short* base, size_t elemBase, int lh) {
  Frag f;
  const unsigned int* p = (const unsigned int*)base;
  size_t db = elemBase >> 1;  // elemBase is always even
#pragma unroll
  for (int j = 0; j < 8; ++j) {
    int pair = j + ((j >> 2) << 2) + (lh << 2);
    f.u[j] = p[db + pair];
  }
  return f;
}

// ---------------------------------------------------------------------------
// Generic 128x128-tile WMMA GEMM: C[M,N] = A[M,K] * B[K,N]
// A: f32 or bf16 (template), B: f32 (converted to bf16 on load), C: f32 or bf16
// Block: 256 threads (8 wave32s), each wave owns a 16-row strip x 128 cols.
// bf16 A tiles are moved with GLOBAL_LOAD_ASYNC_TO_LDS_B128 (ASYNCcnt path).
// ---------------------------------------------------------------------------
template <bool A_BF16, bool OUT_F32>
__global__ __launch_bounds__(256) void gemm_wmma_kernel(const void* __restrict__ Av,
                                                        const float* __restrict__ Bf,
                                                        void* __restrict__ Cv,
                                                        int M, int N, int K) {
  __shared__ __align__(16) unsigned short lds_a[128 * 32];   // [row][k]
  __shared__ __align__(16) unsigned short lds_b[128 * 32];   // [n][k] (transposed)
  int tid = threadIdx.x;
  int ntiles = N >> 7;
  int mbase = (blockIdx.x / ntiles) << 7;
  int nbase = (blockIdx.x % ntiles) << 7;
  int w = tid >> 5, lane = tid & 31, r = lane & 15, lh = lane >> 4;

  const float* Af = (const float*)Av;
  const unsigned short* Ab = (const unsigned short*)Av;

  f32x8 z = {0, 0, 0, 0, 0, 0, 0, 0};
  f32x8 acc[8];
#pragma unroll
  for (int i = 0; i < 8; ++i) acc[i] = z;

  for (int kb = 0; kb < K; kb += 32) {
    {  // A tile [128 rows][32 k], 16 elems (32 B) per thread
      int row = tid >> 1, k0 = (tid & 1) << 4;
      size_t g = (size_t)(mbase + row) * K + kb + k0;
      if (A_BF16) {
        async_copy32(Ab + g, lds_a + row * 32 + k0);   // ASYNCcnt DMA, no VGPRs
      } else {
#pragma unroll
        for (int j = 0; j < 16; ++j) lds_a[row * 32 + k0 + j] = f2bf(Af[g + j]);
      }
    }
    {  // B tile: read [32 k][128 n] row-major, store transposed [n][k]
      int kk = tid >> 3, n0 = (tid & 7) << 4;
      size_t g = (size_t)(kb + kk) * N + nbase + n0;
#pragma unroll
      for (int j = 0; j < 16; ++j) lds_b[(n0 + j) * 32 + kk] = f2bf(Bf[g + j]);
    }
    if (kb + 32 < K)  // next weight tile -> global_prefetch
      __builtin_prefetch(&Bf[(size_t)(kb + 32) * N + nbase + ((tid & 7) << 4)], 0, 0);
    if (A_BF16) wait_async0();  // async tile complete before barrier signal
    __syncthreads();

    Frag a = frag_from(lds_a, (size_t)(w * 16 + r) * 32, lh);
#pragma unroll
    for (int ct = 0; ct < 8; ++ct) {
      Frag bb = frag_from(lds_b, (size_t)(ct * 16 + r) * 32, lh);
      acc[ct] = wmma_bf16(a.v, bb.v, acc[ct]);
    }
    __syncthreads();
  }

#pragma unroll
  for (int ct = 0; ct < 8; ++ct) {
#pragma unroll
    for (int i = 0; i < 8; ++i) {
      int row = mbase + w * 16 + i + (lh << 3);
      int col = nbase + ct * 16 + r;
      if (OUT_F32) ((float*)Cv)[(size_t)row * N + col] = acc[ct][i];
      else ((unsigned short*)Cv)[(size_t)row * N + col] = f2bf(acc[ct][i]);
    }
  }
}

// ---------------------------------------------------------------------------
// Splat preprocessing: normalize positions/directions, fold constants.
// scal layout: [0..127]=pos_sq  [128..255]=dot(pos,dir)  [256..383]=0.5/s^2
//              [384..511]=softmax amplitude
// ---------------------------------------------------------------------------
__global__ void prep_splats_kernel(const float* __restrict__ pos, const float* __restrict__ dir,
                                   const float* __restrict__ ls, const float* __restrict__ la,
                                   float* __restrict__ posn, float* __restrict__ dirn,
                                   float* __restrict__ scal) {
  int t = threadIdx.x;           // t = h*NS + n
  if (t >= NH_ * NS_) return;
  int h = t >> 3;
  const float* p = pos + (size_t)t * D_;
  const float* dd = dir + (size_t)t * D_;
  float sp = 0.0f, sd2 = 0.0f;
  for (int i = 0; i < D_; ++i) { sp += p[i] * p[i]; sd2 += dd[i] * dd[i]; }
  float inp = (sqrtf((float)D_) * 0.3f) / (sqrtf(sp) + 1e-12f);
  float ind = 1.0f / (sqrtf(sd2) + 1e-12f);
  float pd = 0.0f, ps = 0.0f;
  for (int i = 0; i < D_; ++i) {
    float pv = p[i] * inp, dv = dd[i] * ind;
    posn[(size_t)t * D_ + i] = pv;
    dirn[(size_t)t * D_ + i] = dv;
    pd += pv * dv;
    ps += pv * pv;
  }
  float sc = __expf(ls[t]);
  sc = fminf(fmaxf(sc, 0.3f), 1.2f);
  float mx = -1e30f;
  for (int j = 0; j < 8; ++j) mx = fmaxf(mx, la[h * 8 + j]);
  float se = 0.0f;
  for (int j = 0; j < 8; ++j) se += __expf(la[h * 8 + j] - mx);
  float amp = __expf(la[t] - mx) / se;
  scal[t] = ps;
  scal[128 + t] = pd;
  scal[256 + t] = 0.5f / (sc * sc);
  scal[384 + t] = amp;
}

// ---------------------------------------------------------------------------
// Affinities: one wave32 per (b,h,s) token; lane owns 4 of 128 dims.
// aff[b,h,s,n] = (1-ds)*exp(-dist2/2s^2) + ds*exp(-perp2/2s^2)
// ---------------------------------------------------------------------------
__global__ __launch_bounds__(256) void affinity_kernel(
    const unsigned short* __restrict__ Tbf, float* __restrict__ aff,
    const float* __restrict__ posn, const float* __restrict__ dirn,
    const float* __restrict__ scal, const float* __restrict__ dsp) {
  int tid = threadIdx.x;
  int lane = tid & 31;
  int item = blockIdx.x * 8 + (tid >> 5);   // (b*NH + h)*S + s
  int s = item & (S_ - 1);
  int h = (item >> 10) & (NH_ - 1);
  int b = item >> 14;

  float tok[4];
#pragma unroll
  for (int j = 0; j < 4; ++j)
    tok[j] = bf2f(Tbf[((size_t)(b * S_ + s)) * HID_ + h * 128 + lane + 32 * j]);

  float ts = 0.0f, tp[8], td[8];
#pragma unroll
  for (int n = 0; n < 8; ++n) { tp[n] = 0.0f; td[n] = 0.0f; }
#pragma unroll
  for (int j = 0; j < 4; ++j) {
    float x = tok[j];
    ts += x * x;
    int d = lane + 32 * j;
#pragma unroll
    for (int n = 0; n < 8; ++n) {
      tp[n] += x * posn[(h * 8 + n) * 128 + d];
      td[n] += x * dirn[(h * 8 + n) * 128 + d];
    }
  }
#pragma unroll
  for (int msk = 16; msk >= 1; msk >>= 1) {   // full wave32 tree reduce
    ts += __shfl_xor(ts, msk, 32);
#pragma unroll
    for (int n = 0; n < 8; ++n) {
      tp[n] += __shfl_xor(tp[n], msk, 32);
      td[n] += __shfl_xor(td[n], msk, 32);
    }
  }
  if (lane < NS_) {
    int n = lane, t = h * 8 + n;
    float ds = sigmoidf_(dsp[0]);
    float dist2 = fmaxf(ts - 2.0f * tp[n] + scal[t], 0.0f);
    float proj = td[n] - scal[128 + t];
    float perp2 = fmaxf(dist2 - proj * proj, 0.0f);
    float iv = scal[256 + t];
    float a = (1.0f - ds) * __expf(-dist2 * iv) + ds * __expf(-perp2 * iv);
    aff[(((size_t)(b * NH_ + h)) * S_ + s) * NS_ + n] = a;
  }
}

// ---------------------------------------------------------------------------
// Rank-NS GSA reduction: Mamp[b,h,n,d] = amp[h,n] * sum_s ka[s,n]*V[s,d]
//                        Ska [b,h,n]   = amp[h,n] * sum_s ka[s,n]
// ---------------------------------------------------------------------------
__global__ __launch_bounds__(128) void gsa_reduce_kernel(
    const unsigned short* __restrict__ Vbf, const float* __restrict__ ka,
    const float* __restrict__ scal, float* __restrict__ Mamp, float* __restrict__ Ska) {
  int bh = blockIdx.x;
  int b = bh >> 4, h = bh & 15;
  int d = threadIdx.x;
  float acc[8];
#pragma unroll
  for (int n = 0; n < 8; ++n) acc[n] = 0.0f;
  float ss = 0.0f;
  for (int s = 0; s < S_; ++s) {
    float v = bf2f(Vbf[((size_t)(b * S_ + s)) * HID_ + h * 128 + d]);
    const float* kr = ka + (((size_t)bh) * S_ + s) * NS_;
#pragma unroll
    for (int n = 0; n < 8; ++n) acc[n] += kr[n] * v;
    if (d < NS_) ss += kr[d];
  }
#pragma unroll
  for (int n = 0; n < 8; ++n)
    Mamp[((size_t)bh * NS_ + n) * 128 + d] = scal[384 + h * 8 + n] * acc[n];
  if (d < NS_) Ska[bh * NS_ + d] = scal[384 + h * 8 + d] * ss;
}

// ---------------------------------------------------------------------------
// Fused flash attention + GSA blend. Block = 256 thr (8 waves); each wave owns
// a 16-query tile, block covers 128 queries of one (b,h). K tiles are DMA'd
// with GLOBAL_LOAD_ASYNC_TO_LDS_B128; V goes through VGPRs (needs transpose).
// Scores & P never leave registers/LDS.
// ---------------------------------------------------------------------------
__global__ __launch_bounds__(256) void attn_kernel(
    const unsigned short* __restrict__ Qbf, const unsigned short* __restrict__ Kbf,
    const unsigned short* __restrict__ Vbf, const float* __restrict__ qa,
    const float* __restrict__ Mamp, const float* __restrict__ Ska,
    unsigned short* __restrict__ AObf, const float* __restrict__ gsa_p) {
  __shared__ __align__(16) unsigned short lds_k[32 * 128];     // [tok][dim]
  __shared__ __align__(16) unsigned short lds_vt[128 * 32];    // [dim][tok]
  __shared__ __align__(16) unsigned short lds_p[8 * 16 * 32];  // per-wave P staging
  __shared__ float lds_qa[128 * 8];
  __shared__ float lds_m[128 * 8];   // [col][n]
  __shared__ float lds_ska[NS_];

  int tid = threadIdx.x;
  int bh = blockIdx.x >> 3, qblk = blockIdx.x & 7;
  int b = bh >> 4, h = bh & 15;
  int w = tid >> 5, lane = tid & 31, r = lane & 15, lh = lane >> 4;
  int qbase = qblk * 128 + w * 16;
  const float rscale = 0.08838834764831845f;  // 1/sqrt(128)

  // Q fragments straight from global (pairs are contiguous in row-major Q)
  Frag Aq[4];
#pragma unroll
  for (int kd = 0; kd < 4; ++kd)
    Aq[kd] = frag_from(Qbf, ((size_t)(b * S_ + qbase + r)) * HID_ + h * 128 + kd * 32, lh);

  f32x8 z = {0, 0, 0, 0, 0, 0, 0, 0};
  f32x8 O[8];
#pragma unroll
  for (int i = 0; i < 8; ++i) O[i] = z;
  float mrow[8], lrow[8];
#pragma unroll
  for (int i = 0; i < 8; ++i) { mrow[i] = -1e30f; lrow[i] = 0.0f; }

  for (int kt = 0; kt < S_ / 32; ++kt) {
    {  // K tile token-major via async DMA; V tile transposed via VGPRs
      int tok = tid >> 3, d0 = (tid & 7) << 4;
      size_t g = ((size_t)(b * S_ + kt * 32 + tok)) * HID_ + h * 128 + d0;
      async_copy32(Kbf + g, lds_k + tok * 128 + d0);
#pragma unroll
      for (int j = 0; j < 16; ++j) lds_vt[(d0 + j) * 32 + tok] = Vbf[g + j];
    }
    wait_async0();  // own async K slice complete before barrier signal
    __syncthreads();

    // scores: 16 queries x 32 keys  (two 16x16 WMMA accumulators over D=4x32)
    f32x8 s0 = z, s1 = z;
#pragma unroll
    for (int kd = 0; kd < 4; ++kd) {
      Frag b0 = frag_from(lds_k, (size_t)r * 128 + kd * 32, lh);
      s0 = wmma_bf16(Aq[kd].v, b0.v, s0);
      Frag b1 = frag_from(lds_k, (size_t)(16 + r) * 128 + kd * 32, lh);
      s1 = wmma_bf16(Aq[kd].v, b1.v, s1);
    }

    // online softmax per row (row = i + 8*half lives in one half's 16 lanes)
#pragma unroll
    for (int i = 0; i < 8; ++i) {
      float a0 = s0[i] * rscale, a1 = s1[i] * rscale;
      float sm = fmaxf(a0, a1);
#pragma unroll
      for (int msk = 1; msk < 16; msk <<= 1) sm = fmaxf(sm, __shfl_xor(sm, msk, 32));
      float nm = fmaxf(mrow[i], sm);
      float c = __expf(mrow[i] - nm);
      float p0 = __expf(a0 - nm), p1 = __expf(a1 - nm);
      float rs = p0 + p1;
#pragma unroll
      for (int msk = 1; msk < 16; msk <<= 1) rs += __shfl_xor(rs, msk, 32);
      lrow[i] = lrow[i] * c + rs;
      mrow[i] = nm;
#pragma unroll
      for (int ns = 0; ns < 8; ++ns) O[ns][i] *= c;
      int prow = i + (lh << 3);  // stage P for the C->A layout transpose
      lds_p[w * 512 + prow * 32 + r] = f2bf(p0);
      lds_p[w * 512 + prow * 32 + 16 + r] = f2bf(p1);
    }
    asm volatile("s_wait_dscnt 0" ::: "memory");  // wave-local LDS store->load

    Frag P = frag_from(lds_p, (size_t)(w * 512 + r * 32), lh);
#pragma unroll
    for (int ns = 0; ns < 8; ++ns) {
      Frag bv = frag_from(lds_vt, (size_t)(ns * 16 + r) * 32, lh);
      O[ns] = wmma_bf16(P.v, bv.v, O[ns]);
    }
    __syncthreads();
  }

  // epilogue: stage GSA tables, then blend + write bf16 attention output
  {
    size_t qa_base = (((size_t)bh) * S_ + qblk * 128) * NS_;
#pragma unroll
    for (int j = 0; j < 4; ++j) {
      int idx = tid * 4 + j;
      lds_qa[idx] = qa[qa_base + idx];
      int n = idx >> 7, col = idx & 127;
      lds_m[col * 8 + n] = Mamp[((size_t)bh * NS_ + n) * 128 + col];
    }
    if (tid < NS_) lds_ska[tid] = Ska[bh * NS_ + tid];
  }
  __syncthreads();

  float blend = fminf(0.05f, sigmoidf_(gsa_p[0]) * 0.1f);
#pragma unroll
  for (int i = 0; i < 8; ++i) {
    int tl = w * 16 + i + (lh << 3);
    float q8[8];
#pragma unroll
    for (int n = 0; n < 8; ++n) q8[n] = lds_qa[tl * 8 + n];
    float den = 1e-8f;
#pragma unroll
    for (int n = 0; n < 8; ++n) den += q8[n] * lds_ska[n];
    float invl = 1.0f / lrow[i];
    size_t orow = ((size_t)(b * S_ + qblk * 128 + tl)) * HID_ + h * 128;
#pragma unroll
    for (int ns = 0; ns < 8; ++ns) {
      int col = ns * 16 + r;
      float dg = 0.0f;
#pragma unroll
      for (int n = 0; n < 8; ++n) dg += q8[n] * lds_m[col * 8 + n];
      float val = (1.0f - blend) * (O[ns][i] * invl) + blend * (dg / den);
      AObf[orow + col] = f2bf(val);
    }
  }
}

// ---------------------------------------------------------------------------
extern "C" void kernel_launch(void* const* d_in, const int* in_sizes, int n_in,
                              void* d_out, int out_size, void* d_ws, size_t ws_size,
                              hipStream_t stream) {
  (void)in_sizes; (void)n_in; (void)out_size; (void)ws_size;
  const float* X   = (const float*)d_in[0];
  const float* Wq  = (const float*)d_in[1];
  const float* Wk  = (const float*)d_in[2];
  const float* Wv  = (const float*)d_in[3];
  const float* Wo  = (const float*)d_in[4];
  const float* sp  = (const float*)d_in[5];
  const float* sd  = (const float*)d_in[6];
  const float* sls = (const float*)d_in[7];
  const float* sla = (const float*)d_in[8];
  const float* dsp = (const float*)d_in[9];
  const float* gsp = (const float*)d_in[10];

  char* ws = (char*)d_ws;
  size_t tokBytes = (size_t)B_ * S_ * HID_ * 2;               // 8 MB each
  unsigned short* Qbf  = (unsigned short*)(ws + 0 * tokBytes);
  unsigned short* Kbf  = (unsigned short*)(ws + 1 * tokBytes);
  unsigned short* Vbf  = (unsigned short*)(ws + 2 * tokBytes);
  unsigned short* AObf = (unsigned short*)(ws + 3 * tokBytes);
  size_t off = 4 * tokBytes;
  size_t affBytes = (size_t)B_ * NH_ * S_ * NS_ * 4;          // 1 MB each
  float* qa   = (float*)(ws + off); off += affBytes;
  float* ka   = (float*)(ws + off); off += affBytes;
  float* posn = (float*)(ws + off); off += (size_t)NH_ * NS_ * D_ * 4;
  float* dirn = (float*)(ws + off); off += (size_t)NH_ * NS_ * D_ * 4;
  float* scal = (float*)(ws + off); off += (size_t)4 * NH_ * NS_ * 4;
  float* Mamp = (float*)(ws + off); off += (size_t)B_ * NH_ * NS_ * D_ * 4;
  float* Ska  = (float*)(ws + off); off += (size_t)B_ * NH_ * NS_ * 4;

  const int M = B_ * S_;  // 2048

  prep_splats_kernel<<<1, 128, 0, stream>>>(sp, sd, sls, sla, posn, dirn, scal);
  gemm_wmma_kernel<false, false><<<256, 256, 0, stream>>>(X, Wq, Qbf, M, HID_, HID_);
  gemm_wmma_kernel<false, false><<<256, 256, 0, stream>>>(X, Wk, Kbf, M, HID_, HID_);
  gemm_wmma_kernel<false, false><<<256, 256, 0, stream>>>(X, Wv, Vbf, M, HID_, HID_);
  affinity_kernel<<<(B_ * NH_ * S_) / 8, 256, 0, stream>>>(Qbf, qa, posn, dirn, scal, dsp);
  affinity_kernel<<<(B_ * NH_ * S_) / 8, 256, 0, stream>>>(Kbf, ka, posn, dirn, scal, dsp);
  gsa_reduce_kernel<<<B_ * NH_, 128, 0, stream>>>(Vbf, ka, scal, Mamp, Ska);
  attn_kernel<<<B_ * NH_ * (S_ / 128), 256, 0, stream>>>(Qbf, Kbf, Vbf, qa, Mamp, Ska,
                                                         AObf, gsp);
  gemm_wmma_kernel<true, true><<<256, 256, 0, stream>>>(AObf, Wo, d_out, M, HID_, HID_);
}
